// GraphSAGEEncoder_62964220559632
// MI455X (gfx1250) — compile-verified, matrix operand order checked by reading, not run on hardware
//
#include <hip/hip_runtime.h>

#define D 128
#define ROW_TILES 4   // 16-row tiles per block

typedef __attribute__((ext_vector_type(16))) __bf16 v16bf;
typedef __attribute__((ext_vector_type(8)))  float  v8f;

// ---------------------------------------------------------------------------
// float -> bf16 via native hardware convert (v_cvt_pk_bf16_f32)
// ---------------------------------------------------------------------------
__device__ __forceinline__ __bf16 f2bf(float f) { return (__bf16)f; }

// Build a 16x32 bf16 WMMA fragment from a row pointer already offset by
// (row*D + kb + kh*8):  elements 0..7 = p[0..7]*s, elements 8..15 = p[16..23]*s
__device__ __forceinline__ v16bf load_frag(const float* __restrict__ p, float s) {
    const float4* f0 = reinterpret_cast<const float4*>(p);
    const float4* f1 = reinterpret_cast<const float4*>(p + 16);
    float4 q0 = f0[0], q1 = f0[1];
    float4 q2 = f1[0], q3 = f1[1];
    float t[16] = {q0.x, q0.y, q0.z, q0.w, q1.x, q1.y, q1.z, q1.w,
                   q2.x, q2.y, q2.z, q2.w, q3.x, q3.y, q3.z, q3.w};
    v16bf r;
#pragma unroll
    for (int i = 0; i < 16; ++i) r[i] = f2bf(t[i] * s);
    return r;
}

// ---------------------------------------------------------------------------
// Zero the (agg, cnt) workspace — float4 wide
// ---------------------------------------------------------------------------
__global__ void zero_ws(float* __restrict__ p, size_t n) {
    size_t i4 = ((size_t)blockIdx.x * blockDim.x + threadIdx.x) * 4;
    if (i4 + 3 < n) {
        *reinterpret_cast<float4*>(p + i4) = make_float4(0.f, 0.f, 0.f, 0.f);
    } else {
        for (size_t j = i4; j < n; ++j) p[j] = 0.0f;
    }
}

// ---------------------------------------------------------------------------
// Edge scatter: one wave32 per edge; each lane moves 4 contiguous floats.
//   agg[dst] += x[src];  cnt[dst] += 1
// Edge indices are streamed nontemporally; node rows stay cache-resident.
// ---------------------------------------------------------------------------
__global__ __launch_bounds__(256) void sage_scatter(
    const float* __restrict__ x,
    const int* __restrict__ src, const int* __restrict__ dst,
    float* __restrict__ agg, float* __restrict__ cnt, int n_edges) {
    int e = (int)(((size_t)blockIdx.x * blockDim.x + threadIdx.x) >> 5);
    if (e >= n_edges) return;
    int lane = threadIdx.x & 31;
    int s = __builtin_nontemporal_load(src + e);
    int d = __builtin_nontemporal_load(dst + e);
    float4 v = reinterpret_cast<const float4*>(x + (size_t)s * D)[lane];
    float* a = agg + (size_t)d * D + lane * 4;
    atomicAdd(a + 0, v.x);
    atomicAdd(a + 1, v.y);
    atomicAdd(a + 2, v.z);
    atomicAdd(a + 3, v.w);
    if (lane == 0) atomicAdd(cnt + d, 1.0f);
}

// ---------------------------------------------------------------------------
// Fused SAGE linear:  out = (agg/max(cnt,1)) @ Wl^T + x @ Wr^T + b  [+ GELU]
// One block covers ROW_TILES x 16 rows; 8 waves cover the 128 output columns.
// K-block is the OUTER loop: per K-step one {bWl,bWr} fragment pair (16 VGPRs)
// feeds ROW_TILES independent accumulator chains -> deep ILP for the scheduler
// (32 independent b128 loads + 8 WMMAs per K-step, hazard slots filled by
// sibling tiles).  32 WMMAs total per kernel.
// ---------------------------------------------------------------------------
template <bool GELU_ACT>
__global__ __launch_bounds__(256) void sage_gemm(
    const float* __restrict__ agg, const float* __restrict__ cnt,
    const float* __restrict__ xin, const float* __restrict__ wl,
    const float* __restrict__ wr, const float* __restrict__ bias,
    float* __restrict__ out, int n_nodes) {
    const int wave  = threadIdx.x >> 5;    // 0..7 -> column tile
    const int lane  = threadIdx.x & 31;
    const int r     = lane & 15;
    const int kh    = lane >> 4;           // K-half selector
    const int nglob = wave * 16 + r;       // output column == weight row

    const float* wlRow = wl + (size_t)nglob * D + kh * 8;
    const float* wrRow = wr + (size_t)nglob * D + kh * 8;

    const int rowBase = blockIdx.x * (16 * ROW_TILES);

    // per-tile row state (clamped so EXEC stays full for WMMA)
    float        invc[ROW_TILES];
    const float* aggRow[ROW_TILES];
    const float* xRow[ROW_TILES];
#pragma unroll
    for (int t = 0; t < ROW_TILES; ++t) {
        int rowA = rowBase + t * 16 + r;
        if (rowA >= n_nodes) rowA = n_nodes - 1;
        invc[t]   = 1.0f / fmaxf(cnt[rowA], 1.0f);
        aggRow[t] = agg + (size_t)rowA * D + kh * 8;
        xRow[t]   = xin + (size_t)rowA * D + kh * 8;
    }

    v8f acc[ROW_TILES];
#pragma unroll
    for (int t = 0; t < ROW_TILES; ++t) acc[t] = (v8f){};

#pragma unroll
    for (int i = 0; i < 4; ++i) {                  // K-blocks of 32
        const int kb = i * 32;
        const v16bf bWl = load_frag(wlRow + kb, 1.0f);
        const v16bf bWr = load_frag(wrRow + kb, 1.0f);
#pragma unroll
        for (int t = 0; t < ROW_TILES; ++t) {
            v16bf aA = load_frag(aggRow[t] + kb, invc[t]);
            acc[t] = __builtin_amdgcn_wmma_f32_16x16x32_bf16(
                false, aA, false, bWl, (short)0, acc[t], false, false);
            v16bf aX = load_frag(xRow[t] + kb, 1.0f);
            acc[t] = __builtin_amdgcn_wmma_f32_16x16x32_bf16(
                false, aX, false, bWr, (short)0, acc[t], false, false);
        }
    }

    const float bn    = bias[nglob];
    const int   mbase = kh * 8;
#pragma unroll
    for (int t = 0; t < ROW_TILES; ++t) {
#pragma unroll
        for (int g = 0; g < 8; ++g) {
            int rowOut = rowBase + t * 16 + mbase + g;
            if (rowOut < n_nodes) {
                float v = acc[t][g] + bn;
                if (GELU_ACT) v = 0.5f * v * (1.0f + erff(v * 0.70710678118654752f));
                out[(size_t)rowOut * D + nglob] = v;
            }
        }
    }
}

// ---------------------------------------------------------------------------
extern "C" void kernel_launch(void* const* d_in, const int* in_sizes, int n_in,
                              void* d_out, int out_size, void* d_ws, size_t ws_size,
                              hipStream_t stream) {
    const float* embs = (const float*)d_in[0];
    const int*   ei0  = (const int*)d_in[1];   // (2,E): [src | dst]
    const int*   ei1  = (const int*)d_in[2];
    const float* wl0  = (const float*)d_in[3];
    const float* wr0  = (const float*)d_in[4];
    const float* b0   = (const float*)d_in[5];
    const float* wl1  = (const float*)d_in[6];
    const float* wr1  = (const float*)d_in[7];
    const float* b1   = (const float*)d_in[8];

    const int N = in_sizes[0] / D;
    const int E = in_sizes[1] / 2;

    float* agg = (float*)d_ws;                 // N*D floats
    float* cnt = agg + (size_t)N * D;          // N floats (contiguous -> zero together)
    float* h   = (float*)d_out;                // layer-0 activations staged in d_out

    const size_t zn = (size_t)N * D + (size_t)N;
    const int zb = (int)((zn / 4 + 255) / 256) + 1;
    const int sb = (E + 7) / 8;                // 8 edges (waves) per 256-thread block
    const int gb = (N + 16 * ROW_TILES - 1) / (16 * ROW_TILES);

    // ---- layer 0: agg = scatter-mean(embs); h = gelu(agg@Wl0^T + embs@Wr0^T + b0)
    zero_ws<<<zb, 256, 0, stream>>>(agg, zn);
    sage_scatter<<<sb, 256, 0, stream>>>(embs, ei0, ei0 + E, agg, cnt, E);
    sage_gemm<true><<<gb, 256, 0, stream>>>(agg, cnt, embs, wl0, wr0, b0, h, N);

    // ---- layer 1: agg = scatter-mean(h); out = agg@Wl1^T + h@Wr1^T + b1
    zero_ws<<<zb, 256, 0, stream>>>(agg, zn);
    sage_scatter<<<sb, 256, 0, stream>>>(h, ei1, ei1 + E, agg, cnt, E);
    sage_gemm<false><<<gb, 256, 0, stream>>>(agg, cnt, h, wl1, wr1, b1,
                                             (float*)d_out, N);
}
